// GraphSAGEModel_82325933130195
// MI455X (gfx1250) — matrix operand order repaired and checked by reading, not compiled, not run on hardware
//
#include <hip/hip_runtime.h>

typedef __attribute__((ext_vector_type(16))) _Float16 v16h;
typedef __attribute__((ext_vector_type(8)))  float    v8f;

#define N_NODES 50000
#define N_EDGES 800000
#define DHID    128
#define DOUT    64
#define NSTRIPS (N_NODES / 16)   // 3125 row-strips of 16

// ---------------------------------------------------------------- utilities
__global__ void zero_f32_kernel(float* __restrict__ p, int n) {
    int i = blockIdx.x * blockDim.x + threadIdx.x;
    if (i < n) p[i] = 0.0f;
}

// degree accumulation (float so we can reuse atomicAdd f32)
__global__ void deg_kernel(const long long* __restrict__ tgt, float* __restrict__ deg) {
    int e = blockIdx.x * blockDim.x + threadIdx.x;
    if (e < N_EDGES) atomicAdd(&deg[(int)tgt[e]], 1.0f);
}

__global__ void invdeg_kernel(float* __restrict__ deg) {
    int i = blockIdx.x * blockDim.x + threadIdx.x;
    if (i < N_NODES) deg[i] = 1.0f / fmaxf(deg[i], 1.0f);
}

// ---------------------------------------------------------------- weight packing
// Pack row-major f32 W[128][128] into WMMA-B f16 tiles.
// Tile (kt, nt): 32x16 block, B layout (16-bit, 32x16, wave32):
//   lanes 0-15 : N = lane,    elem e holds K = kt*32 + (e&7) + (e>>3)*16
//   lanes 16-31: N = lane-16, elem e holds K = kt*32 + 8 + (e&7) + (e>>3)*16
// Stored so a lane reads its v16h with one contiguous 32-byte load:
//   P[((kt*8 + nt)*32 + lane)*16 + elem],  K = kt*32 + (lane>>4)*8 + (elem&7) + (elem>>3)*16
__global__ void pack_w_kernel(const float* __restrict__ W, _Float16* __restrict__ P) {
    int idx = blockIdx.x * blockDim.x + threadIdx.x;   // 4*8*512 = 16384 halves
    if (idx >= 4 * 8 * 512) return;
    int t    = idx >> 9;        // tile id = kt*8 + nt
    int r    = idx & 511;
    int lane = r >> 4;
    int elem = r & 15;
    int kt = t >> 3, nt = t & 7;
    int K    = kt * 32 + (lane >> 4) * 8 + (elem & 7) + (elem >> 3) * 16;
    int Ncol = nt * 16 + (lane & 15);
    P[idx] = (_Float16)W[K * DHID + Ncol];
}

// ---------------------------------------------------------------- aggregation
// one wave32 per edge; lane moves 4 consecutive floats (32 lanes * 4 = 128)
__global__ void scatter_kernel(const float* __restrict__ h,
                               const long long* __restrict__ src,
                               const long long* __restrict__ tgt,
                               float* __restrict__ agg) {
    int t    = blockIdx.x * blockDim.x + threadIdx.x;
    int e    = t >> 5;
    int lane = t & 31;
    if (e >= N_EDGES) return;
    int s = (int)src[e];
    int d = (int)tgt[e];
    const float4 v = *(const float4*)(h + (size_t)s * DHID + lane * 4);
    float* dst = agg + (size_t)d * DHID + lane * 4;
    atomicAdd(dst + 0, v.x);
    atomicAdd(dst + 1, v.y);
    atomicAdd(dst + 2, v.z);
    atomicAdd(dst + 3, v.w);
}

// ---------------------------------------------------------------- fused SAGE layer GEMM
// out[i,:] = relu( h[i,:] @ Ws + (agg[i,:]*invdeg[i]) @ Wn + b )
// One wave per 16-row strip; 8 column tiles of 16; K loop in 4 steps of 32.
// A layout (16-bit A 16x32): lane M = lane&15, half = lane>>4,
//   elems 0..7  -> K = kk + half*8 + j
//   elems 8..15 -> K = kk + 16 + half*8 + j
__global__ void __launch_bounds__(128)
sage_gemm_kernel(const float* __restrict__ h, const float* __restrict__ agg,
                 const float* __restrict__ invdeg,
                 const _Float16* __restrict__ Pws, const _Float16* __restrict__ Pwn,
                 const float* __restrict__ bias, float* __restrict__ out) {
    int wave = (blockIdx.x * blockDim.x + threadIdx.x) >> 5;
    int lane = threadIdx.x & 31;
    if (wave >= NSTRIPS) return;           // uniform per wave -> EXEC all-1 inside

    const int i0   = wave * 16;
    const int M    = lane & 15;
    const int half = lane >> 4;
    const int row  = i0 + M;
    const float idg = invdeg[row];

    v8f acc[8] = {};

    #pragma unroll
    for (int kt = 0; kt < 4; ++kt) {
        const int kk = kt * 32;
        const float* ph = h   + (size_t)row * DHID + kk + half * 8;
        const float* pa = agg + (size_t)row * DHID + kk + half * 8;

        float4 h0 = *(const float4*)(ph);
        float4 h1 = *(const float4*)(ph + 4);
        float4 h2 = *(const float4*)(ph + 16);
        float4 h3 = *(const float4*)(ph + 20);
        float4 a0 = *(const float4*)(pa);
        float4 a1 = *(const float4*)(pa + 4);
        float4 a2 = *(const float4*)(pa + 16);
        float4 a3 = *(const float4*)(pa + 20);

        v16h ah, an;
        ah[0]  = (_Float16)h0.x; ah[1]  = (_Float16)h0.y; ah[2]  = (_Float16)h0.z; ah[3]  = (_Float16)h0.w;
        ah[4]  = (_Float16)h1.x; ah[5]  = (_Float16)h1.y; ah[6]  = (_Float16)h1.z; ah[7]  = (_Float16)h1.w;
        ah[8]  = (_Float16)h2.x; ah[9]  = (_Float16)h2.y; ah[10] = (_Float16)h2.z; ah[11] = (_Float16)h2.w;
        ah[12] = (_Float16)h3.x; ah[13] = (_Float16)h3.y; ah[14] = (_Float16)h3.z; ah[15] = (_Float16)h3.w;
        an[0]  = (_Float16)(a0.x*idg); an[1]  = (_Float16)(a0.y*idg); an[2]  = (_Float16)(a0.z*idg); an[3]  = (_Float16)(a0.w*idg);
        an[4]  = (_Float16)(a1.x*idg); an[5]  = (_Float16)(a1.y*idg); an[6]  = (_Float16)(a1.z*idg); an[7]  = (_Float16)(a1.w*idg);
        an[8]  = (_Float16)(a2.x*idg); an[9]  = (_Float16)(a2.y*idg); an[10] = (_Float16)(a2.z*idg); an[11] = (_Float16)(a2.w*idg);
        an[12] = (_Float16)(a3.x*idg); an[13] = (_Float16)(a3.y*idg); an[14] = (_Float16)(a3.z*idg); an[15] = (_Float16)(a3.w*idg);

        #pragma unroll
        for (int nt = 0; nt < 8; ++nt) {
            v16h bs = *(const v16h*)(Pws + ((size_t)(kt * 8 + nt) * 32 + lane) * 16);
            acc[nt] = __builtin_amdgcn_wmma_f32_16x16x32_f16(
                false, ah, false, bs, (short)0, acc[nt], false, false);
            v16h bn = *(const v16h*)(Pwn + ((size_t)(kt * 8 + nt) * 32 + lane) * 16);
            acc[nt] = __builtin_amdgcn_wmma_f32_16x16x32_f16(
                false, an, false, bn, (short)0, acc[nt], false, false);
        }
    }

    // epilogue: C/D layout -> VGPR r: M = r + half*8 ; N = nt*16 + (lane&15)
    #pragma unroll
    for (int nt = 0; nt < 8; ++nt) {
        const int col = nt * 16 + (lane & 15);
        const float b = bias[col];
        #pragma unroll
        for (int r = 0; r < 8; ++r) {
            const int m = r + half * 8;
            float v = acc[nt][r] + b;
            v = fmaxf(v, 0.0f);                      // all 4 layers are followed by relu
            out[(size_t)(i0 + m) * DHID + col] = v;
        }
    }
}

// ---------------------------------------------------------------- mean pool + classify
__global__ void pool_kernel(const float* __restrict__ h, float* __restrict__ g) {
    int c  = threadIdx.x;          // 128 threads = columns
    int r0 = blockIdx.x * 64;
    int r1 = r0 + 64; if (r1 > N_NODES) r1 = N_NODES;
    float s = 0.0f;
    for (int r = r0; r < r1; ++r) s += h[(size_t)r * DHID + c];
    atomicAdd(&g[c], s);
}

__global__ void cls_kernel(const float* __restrict__ g, const float* __restrict__ Wc,
                           const float* __restrict__ bc, float* __restrict__ out) {
    int j = threadIdx.x;
    if (j >= DOUT) return;
    float s = bc[j];
    const float invn = 1.0f / (float)N_NODES;
    for (int k = 0; k < DHID; ++k) s += (g[k] * invn) * Wc[k * DOUT + j];
    out[j] = s;
}

// ---------------------------------------------------------------- launch
extern "C" void kernel_launch(void* const* d_in, const int* in_sizes, int n_in,
                              void* d_out, int out_size, void* d_ws, size_t ws_size,
                              hipStream_t stream) {
    (void)in_sizes; (void)n_in; (void)out_size; (void)ws_size;

    const float*     x   = (const float*)d_in[0];
    const long long* ei  = (const long long*)d_in[1];        // int64 [2, E]
    const long long* src = ei;
    const long long* tgt = ei + N_EDGES;
    const float* W[8] = { (const float*)d_in[2],  (const float*)d_in[3],   // Ws11, Wn11
                          (const float*)d_in[5],  (const float*)d_in[6],   // Ws12, Wn12
                          (const float*)d_in[8],  (const float*)d_in[9],   // Ws21, Wn21
                          (const float*)d_in[11], (const float*)d_in[12] };// Ws22, Wn22
    const float* bias[4] = { (const float*)d_in[4], (const float*)d_in[7],
                             (const float*)d_in[10], (const float*)d_in[13] };
    const float* Wc = (const float*)d_in[14];
    const float* bc = (const float*)d_in[15];
    float* out = (float*)d_out;

    // scratch carve-out
    char* ws = (char*)d_ws;
    size_t off = 0;
    auto carve = [&](size_t bytes) -> void* {
        void* p = ws + off;
        off += (bytes + 511) & ~(size_t)511;
        return p;
    };
    float* hA     = (float*)carve(sizeof(float) * (size_t)N_NODES * DHID);
    float* hB     = (float*)carve(sizeof(float) * (size_t)N_NODES * DHID);
    float* agg    = (float*)carve(sizeof(float) * (size_t)N_NODES * DHID);
    float* invdeg = (float*)carve(sizeof(float) * N_NODES);
    float* g      = (float*)carve(sizeof(float) * DHID);
    _Float16* P[8];
    for (int i = 0; i < 8; ++i) P[i] = (_Float16*)carve(sizeof(_Float16) * 4 * 8 * 512);

    const int T = 256;

    // pack all 8 hidden-layer weight matrices to WMMA-B f16 layout
    for (int i = 0; i < 8; ++i)
        pack_w_kernel<<<(4 * 8 * 512 + T - 1) / T, T, 0, stream>>>(W[i], P[i]);

    // degree -> 1/max(deg,1), computed once (same graph every layer)
    zero_f32_kernel<<<(N_NODES + T - 1) / T, T, 0, stream>>>(invdeg, N_NODES);
    deg_kernel<<<(N_EDGES + T - 1) / T, T, 0, stream>>>(tgt, invdeg);
    invdeg_kernel<<<(N_NODES + T - 1) / T, T, 0, stream>>>(invdeg);

    const int nelem        = N_NODES * DHID;
    const int zero_blocks  = (nelem + T - 1) / T;
    const int scat_blocks  = (N_EDGES * 32 + T - 1) / T;
    const int gemm_blocks  = (NSTRIPS * 32 + 127) / 128;

    auto run_layer = [&](const float* hin, int li, float* hout) {
        zero_f32_kernel<<<zero_blocks, T, 0, stream>>>(agg, nelem);
        scatter_kernel<<<scat_blocks, T, 0, stream>>>(hin, src, tgt, agg);
        sage_gemm_kernel<<<gemm_blocks, 128, 0, stream>>>(
            hin, agg, invdeg, P[2 * li], P[2 * li + 1], bias[li], hout);
    };

    run_layer(x,  0, hA);   // relu(SAGE(x))
    run_layer(hA, 1, hB);   // relu(SAGE(h))
    run_layer(hB, 2, hA);   // relu(SAGE(h))
    run_layer(hA, 3, hB);   // relu(SAGE(h))

    // global mean pool + classifier
    zero_f32_kernel<<<1, DHID, 0, stream>>>(g, DHID);
    pool_kernel<<<(N_NODES + 63) / 64, DHID, 0, stream>>>(hB, g);
    cls_kernel<<<1, 64, 0, stream>>>(g, Wc, bc, out);
}
